// CausalSlidingWindowAttention_53266184405822
// MI455X (gfx1250) — matrix operand order repaired and checked by reading, not compiled, not run on hardware
//
#include <hip/hip_runtime.h>
#include <hip/hip_bf16.h>
#include <stdint.h>

// ---------------------------------------------------------------------------
// CDNA5 (gfx1250) sliding-window causal attention, bf16 WMMA everywhere.
// Round 4: keep round-3 WMMA kernels (pipelined, spill-free); fix the
// preprocessing bandwidth: LDS-tiled transposes (coalesced both sides) and
// vectorized fp32->bf16 conversion.
// ---------------------------------------------------------------------------

typedef __attribute__((ext_vector_type(16))) __bf16 v16bf;
typedef __attribute__((ext_vector_type(8)))  float  v8f;

#define WMMA_BF16(a, b, c) \
  __builtin_amdgcn_wmma_f32_16x16x32_bf16(false, (a), false, (b), (short)0, (c), false, false)

union Frag16 {
  v16bf v;
  uint4 q[2];
};

__device__ __forceinline__ uint16_t f2bf(float f) {
  uint32_t u = __float_as_uint(f);
  uint32_t r = u + 0x7FFFu + ((u >> 16) & 1u);   // round-to-nearest-even
  return (uint16_t)(r >> 16);
}

__device__ __forceinline__ v8f zero8() {
  v8f z = {0.f, 0.f, 0.f, 0.f, 0.f, 0.f, 0.f, 0.f};
  return z;
}

// problem constants
#define BATCH 2
#define TSEQ  2048
#define DMODEL 1024
#define NHEAD 16
#define HD 64
#define WIN 512

// ---------------------------------------------------------------------------
// vectorized elementwise fp32 -> bf16 (8 elements / thread; n % 8 == 0)
__global__ void k_cvt8(const float* __restrict__ src, uint16_t* __restrict__ dst, int n) {
  int i = (blockIdx.x * blockDim.x + threadIdx.x) * 8;
  if (i >= n) return;
  float4 f0 = *(const float4*)(src + i);
  float4 f1 = *(const float4*)(src + i + 4);
  union { uint16_t h[8]; uint4 q; } u;
  u.h[0] = f2bf(f0.x); u.h[1] = f2bf(f0.y); u.h[2] = f2bf(f0.z); u.h[3] = f2bf(f0.w);
  u.h[4] = f2bf(f1.x); u.h[5] = f2bf(f1.y); u.h[6] = f2bf(f1.z); u.h[7] = f2bf(f1.w);
  *(uint4*)(dst + i) = u.q;
}

// W[K x N] row-major fp32 -> WT[N x K] row-major bf16, 32x32 LDS tile,
// coalesced reads (along n) and coalesced writes (along k).
// grid = (N/32, K/32), block = (32, 8)
__global__ void k_transpose_bf16(const float* __restrict__ W, uint16_t* __restrict__ WT,
                                 int K, int N) {
  __shared__ uint16_t tile[32][33];
  const int n0 = blockIdx.x * 32;
  const int k0 = blockIdx.y * 32;
  const int tx = threadIdx.x, ty = threadIdx.y;
#pragma unroll
  for (int i = ty; i < 32; i += 8)
    tile[i][tx] = f2bf(W[(size_t)(k0 + i) * N + n0 + tx]);
  __syncthreads();
#pragma unroll
  for (int i = ty; i < 32; i += 8)
    WT[(size_t)(n0 + i) * K + k0 + tx] = tile[tx][i];
}

// V slice of qkv -> VT[b][h][d][t] (t-contiguous), tiled per (b,h):
// reads coalesced along d, writes coalesced along t.
// grid = (T/32, HD/32, B*H), block = (32, 8)
__global__ void k_build_vt(const uint16_t* __restrict__ qkvb, uint16_t* __restrict__ vT) {
  __shared__ uint16_t tile[32][33];
  const int t0 = blockIdx.x * 32;
  const int d0 = blockIdx.y * 32;
  const int bh = blockIdx.z;
  const int b = bh >> 4, h = bh & 15;
  const int tx = threadIdx.x, ty = threadIdx.y;
  const uint16_t* src = qkvb + (size_t)(b * TSEQ) * (3 * DMODEL) + 2 * DMODEL + h * HD;
#pragma unroll
  for (int i = ty; i < 32; i += 8)
    tile[i][tx] = src[(size_t)(t0 + i) * (3 * DMODEL) + d0 + tx];
  __syncthreads();
  uint16_t* dst = vT + (size_t)bh * HD * TSEQ;
#pragma unroll
  for (int i = ty; i < 32; i += 8)
    dst[(size_t)(d0 + i) * TSEQ + t0 + tx] = tile[tx][i];
}

// ---------------------------------------------------------------------------
// GEMM: C[M x N] = A[M x K] (bf16 row-major) * B (given as BT[N x K] bf16).
// 128 threads = 4 waves; each wave owns a 64x64 tile of a 256x64 block tile.
// Manual unroll-by-2 double buffering with NAMED register sets (K % 64 == 0).
template <bool OUT_BF16>
__global__ void __launch_bounds__(128, 1)
k_gemm(const uint16_t* __restrict__ A, const uint16_t* __restrict__ BT,
       void* __restrict__ Cout, int M, int N, int K) {
  const int lane = threadIdx.x & 31;
  const int wid  = threadIdx.x >> 5;
  const int lr = lane & 15;
  const int g  = lane >> 4;
  const int m0 = blockIdx.y * 256 + wid * 64;
  const int n0 = blockIdx.x * 64;

  const uint16_t* Arow[4];
  const uint16_t* Brow[4];
#pragma unroll
  for (int i = 0; i < 4; ++i) {
    Arow[i] = A  + (size_t)(m0 + 16 * i + lr) * K;
    Brow[i] = BT + (size_t)(n0 + 16 * i + lr) * K;
  }

  v8f acc[4][4];
#pragma unroll
  for (int i = 0; i < 4; ++i)
#pragma unroll
    for (int j = 0; j < 4; ++j) acc[i][j] = zero8();

  Frag16 a0[4], b0[4], a1[4], b1[4];

  auto load_set = [&](Frag16 (&a)[4], Frag16 (&b)[4], int k0) {
#pragma unroll
    for (int i = 0; i < 4; ++i) {
      const uint16_t* p = Arow[i] + k0 + 8 * g;
      a[i].q[0] = *(const uint4*)p;         // K = k0+8g .. +7
      a[i].q[1] = *(const uint4*)(p + 16);  // K = k0+8g+16 .. +23
    }
#pragma unroll
    for (int i = 0; i < 4; ++i) {
      const uint16_t* p = Brow[i] + k0 + 8 * g;
      b[i].q[0] = *(const uint4*)p;         // K = k0+8g .. +7
      b[i].q[1] = *(const uint4*)(p + 16);  // K = k0+8g+16 .. +23
    }
  };

  auto mma_set = [&](Frag16 (&a)[4], Frag16 (&b)[4]) {
#pragma unroll
    for (int am = 0; am < 4; ++am)
#pragma unroll
      for (int bn = 0; bn < 4; ++bn)
        acc[am][bn] = WMMA_BF16(a[am].v, b[bn].v, acc[am][bn]);
  };

  load_set(a0, b0, 0);
  for (int k0 = 0; k0 < K; k0 += 64) {
    load_set(a1, b1, k0 + 32);
    if (k0 + 64 < K) {                      // gfx1250 global_prefetch_b8
#pragma unroll
      for (int i = 0; i < 4; ++i) {
        __builtin_prefetch(Arow[i] + k0 + 64, 0, 1);
        __builtin_prefetch(Brow[i] + k0 + 64, 0, 1);
      }
    }
    mma_set(a0, b0);
    if (k0 + 64 < K) load_set(a0, b0, k0 + 64);
    mma_set(a1, b1);
  }

#pragma unroll
  for (int am = 0; am < 4; ++am)
#pragma unroll
    for (int bn = 0; bn < 4; ++bn)
#pragma unroll
      for (int r = 0; r < 8; ++r) {
        int row = m0 + 16 * am + r + 8 * g;
        int col = n0 + 16 * bn + lr;
        float v = acc[am][bn][r];
        if (OUT_BF16)
          ((uint16_t*)Cout)[(size_t)row * N + col] = f2bf(v);
        else
          ((float*)Cout)[(size_t)row * N + col] = v;
      }
}

// ---------------------------------------------------------------------------
// Fused sliding-window causal flash attention.
// 128 threads = 4 waves; each wave owns one 16-query tile (independent trip
// counts -> NO block barriers; per-wave LDS region + threadfence_block only).
// Single-buffered, fully batched loads: all 4 K frags + 4 V frags load into
// distinct registers (constant indices), so loads clause and wait once.
__global__ void __launch_bounds__(128, 1)
k_attn(const uint16_t* __restrict__ qkvb, const uint16_t* __restrict__ vT,
       uint16_t* __restrict__ attnb) {
  __shared__ __align__(16) uint16_t Plds[4][16][32];

  const int lane = threadIdx.x & 31;
  const int wid  = threadIdx.x >> 5;
  const int lr = lane & 15;
  const int g  = lane >> 4;

  int tile = blockIdx.x * 4 + wid;    // B*H*(T/16) = 4096 tiles
  int qt = tile & (TSEQ / 16 - 1);
  int h  = (tile >> 7) & (NHEAD - 1);
  int b  = tile >> 11;
  int q0 = qt * 16;

  const size_t rs = 3 * DMODEL;       // qkv row stride (elements)
  const uint16_t* Qbase = qkvb + (size_t)(b * TSEQ) * rs + h * HD;
  const uint16_t* Kbase = Qbase + DMODEL;          // k block starts at col D
  const uint16_t* Vtb   = vT + (size_t)((b * NHEAD + h) * HD) * TSEQ;

  // Q A-operand fragments (hd=64 -> two K chunks), held for the whole loop
  Frag16 qa[2];
#pragma unroll
  for (int c = 0; c < 2; ++c) {
    const uint16_t* p = Qbase + (size_t)(q0 + lr) * rs + c * 32 + 8 * g;
    qa[c].q[0] = *(const uint4*)p;
    qa[c].q[1] = *(const uint4*)(p + 16);
  }

  v8f o[4];
#pragma unroll
  for (int dt = 0; dt < 4; ++dt) o[dt] = zero8();
  float mrow[8], lrow[8];
#pragma unroll
  for (int r = 0; r < 8; ++r) { mrow[r] = -1e30f; lrow[r] = 0.f; }

  int kb0 = q0 - WIN + 1;
  if (kb0 < 0) kb0 = 0;
  kb0 &= ~31;
  const int kb_end = q0 + 15;

  for (int kb = kb0; kb <= kb_end; kb += 32) {
    // ---- batched operand loads: 4 K frags then 4 V frags (one clause) ----
    Frag16 kf[2][2];                  // [kt][c] constant-indexed only
    Frag16 vf[4];                     // [dt]    constant-indexed only
#pragma unroll
    for (int kt = 0; kt < 2; ++kt)
#pragma unroll
      for (int c = 0; c < 2; ++c) {
        const uint16_t* p = Kbase + (size_t)(kb + 16 * kt + lr) * rs + c * 32 + 8 * g;
        kf[kt][c].q[0] = *(const uint4*)p;
        kf[kt][c].q[1] = *(const uint4*)(p + 16);
      }
#pragma unroll
    for (int dt = 0; dt < 4; ++dt) {
      const uint16_t* p = Vtb + (size_t)(16 * dt + lr) * TSEQ + kb + 8 * g;
      vf[dt].q[0] = *(const uint4*)p;
      vf[dt].q[1] = *(const uint4*)(p + 16);
    }
    if (kb + 32 <= kb_end) {                       // gfx1250 global_prefetch_b8
      __builtin_prefetch(Kbase + (size_t)(kb + 32 + lr) * rs, 0, 1);
      __builtin_prefetch(Kbase + (size_t)(kb + 48 + lr) * rs, 0, 1);
      __builtin_prefetch(Vtb + (size_t)lr * TSEQ + kb + 32, 0, 1);
    }

    // ---- scores S[16q x 32k] = Q K^T ----
    v8f s[2];
    s[0] = zero8(); s[1] = zero8();
#pragma unroll
    for (int kt = 0; kt < 2; ++kt)
#pragma unroll
      for (int c = 0; c < 2; ++c)
        s[kt] = WMMA_BF16(qa[c].v, kf[kt][c].v, s[kt]);

    // ---- banded mask + online softmax (row = q0+r+8g, col = kb+16kt+lr) ----
    float p0[8], p1[8], al[8];
    const int d0 = q0 + 8 * g - kb - lr;           // i - j0 = d0 + r
#pragma unroll
    for (int r = 0; r < 8; ++r) {
      float s0 = s[0][r] * 0.125f;                 // 1/sqrt(64)
      float s1 = s[1][r] * 0.125f;
      if ((unsigned)(d0 + r) >= WIN)      s0 = -1e30f;   // outside [0, WIN)
      if ((unsigned)(d0 + r - 16) >= WIN) s1 = -1e30f;
      float mx = fmaxf(s0, s1);
#pragma unroll
      for (int d = 1; d < 16; d <<= 1) mx = fmaxf(mx, __shfl_xor(mx, d, 32));
      float mn = fmaxf(mrow[r], mx);
      float e0 = (s0 > -1e29f) ? __expf(s0 - mn) : 0.f;
      float e1 = (s1 > -1e29f) ? __expf(s1 - mn) : 0.f;
      float sum = e0 + e1;
#pragma unroll
      for (int d = 1; d < 16; d <<= 1) sum += __shfl_xor(sum, d, 32);
      float alpha = __expf(mrow[r] - mn);
      lrow[r] = lrow[r] * alpha + sum;
      mrow[r] = mn;
      al[r] = alpha;
      p0[r] = e0;
      p1[r] = e1;
    }
#pragma unroll
    for (int dt = 0; dt < 4; ++dt)
#pragma unroll
      for (int r = 0; r < 8; ++r) o[dt][r] *= al[r];

    // ---- P (C/D layout) -> per-wave LDS -> A-operand layout ----
    // Same-wave LDS ops are in-order (DScnt); fence stops compiler reordering.
#pragma unroll
    for (int r = 0; r < 8; ++r) {
      Plds[wid][r + 8 * g][lr]      = f2bf(p0[r]);
      Plds[wid][r + 8 * g][16 + lr] = f2bf(p1[r]);
    }
    __threadfence_block();
    Frag16 pa;
    {
      const uint16_t* p = &Plds[wid][lr][8 * g];
      pa.q[0] = *(const uint4*)p;
      pa.q[1] = *(const uint4*)(p + 16);
    }
    __threadfence_block();

    // ---- O += P V  (V^T is t-contiguous: perfect B-operand) ----
#pragma unroll
    for (int dt = 0; dt < 4; ++dt)
      o[dt] = WMMA_BF16(pa.v, vf[dt].v, o[dt]);
  }

  // ---- normalize + store bf16 attention output [B*T, D] ----
#pragma unroll
  for (int dt = 0; dt < 4; ++dt)
#pragma unroll
    for (int r = 0; r < 8; ++r) {
      int t = q0 + r + 8 * g;
      float v = o[dt][r] / lrow[r];
      attnb[(size_t)(b * TSEQ + t) * DMODEL + h * HD + 16 * dt + lr] = f2bf(v);
    }
}

// ---------------------------------------------------------------------------
extern "C" void kernel_launch(void* const* d_in, const int* in_sizes, int n_in,
                              void* d_out, int out_size, void* d_ws, size_t ws_size,
                              hipStream_t stream) {
  (void)in_sizes; (void)n_in; (void)out_size; (void)ws_size;
  const float* x     = (const float*)d_in[0];   // [B,T,D]
  const float* w_qkv = (const float*)d_in[1];   // [D, 3D]
  const float* w_out = (const float*)d_in[2];   // [D, D]
  float* out = (float*)d_out;                   // [B,T,D] fp32

  const int M  = BATCH * TSEQ;                  // 4096
  const int N1 = 3 * DMODEL;                    // 3072

  uint16_t* xb    = (uint16_t*)d_ws;                       // 4096*1024
  uint16_t* wqkvT = xb    + (size_t)M * DMODEL;            // 3072*1024
  uint16_t* woutT = wqkvT + (size_t)N1 * DMODEL;           // 1024*1024
  uint16_t* qkvb  = woutT + (size_t)DMODEL * DMODEL;       // 4096*3072
  uint16_t* vT    = qkvb  + (size_t)M * N1;                // 4096*1024
  uint16_t* attnb = vT    + (size_t)M * DMODEL;            // 4096*1024

  int nx = M * DMODEL;
  k_cvt8<<<(nx / 8 + 255) / 256, 256, 0, stream>>>(x, xb, nx);

  dim3 tb(32, 8);
  k_transpose_bf16<<<dim3(N1 / 32, DMODEL / 32), tb, 0, stream>>>(w_qkv, wqkvT, DMODEL, N1);
  k_transpose_bf16<<<dim3(DMODEL / 32, DMODEL / 32), tb, 0, stream>>>(w_out, woutT, DMODEL, DMODEL);

  dim3 g1(N1 / 64, M / 256);
  k_gemm<true><<<g1, 128, 0, stream>>>(xb, wqkvT, qkvb, M, N1, DMODEL);

  k_build_vt<<<dim3(TSEQ / 32, HD / 32, BATCH * NHEAD), tb, 0, stream>>>(qkvb, vT);

  k_attn<<<BATCH * NHEAD * (TSEQ / 16) / 4, 128, 0, stream>>>(qkvb, vT, attnb);

  dim3 g2(DMODEL / 64, M / 256);
  k_gemm<false><<<g2, 128, 0, stream>>>(attnb, woutT, out, M, DMODEL, DMODEL);
}
